// PoolHiddenNet_g_14568529068507
// MI455X (gfx1250) — matrix operand order repaired and checked by reading, not compiled, not run on hardware
//
#include <hip/hip_runtime.h>
#include <hip/hip_bf16.h>

typedef __attribute__((ext_vector_type(16))) __bf16 v16bf;
typedef __attribute__((ext_vector_type(8)))  float  v8f;

#define S_    32
#define P_    64
#define PP    4096
#define ROWS  131072   /* S_*PP */
#define OBS   8
#define EMB   64
#define HDIM  64
#define PRE   576
#define D1    512
#define D2    1024
#define EPSBN 1e-5f

union FragU { v16bf v; unsigned u[8]; };
union Pack4 { uint2 u2; __bf16 h[4]; };
union Pack8 { uint4 u4; __bf16 h[8]; };

// ---------------------------------------------------------------------------
// Convert f32 weights to bf16 AND transpose [K][N] -> [N][K] so that GEMM
// B-tile staging is a pure vectorized copy (K-contiguous rows per n).
// ---------------------------------------------------------------------------
__global__ __launch_bounds__(256) void cvt_wt_kernel(const float* __restrict__ src,
                                                     __bf16* __restrict__ dst,
                                                     int K, int N) {
  int idx = blockIdx.x * 256 + threadIdx.x;
  if (idx < K * N) {
    int n = idx / K, k = idx - n * K;
    dst[idx] = (__bf16)src[k * N + n];
  }
}

// ---------------------------------------------------------------------------
// Build X[131072, 576] (bf16): cols 0..511 = tw * (rel @ W_sp + b_sp),
// cols 512..575 = h[s, j, :].  One workgroup per (s, i) pair.
// ---------------------------------------------------------------------------
__global__ __launch_bounds__(256) void build_x_kernel(
    const float* __restrict__ traj, const float* __restrict__ traj_weight,
    const float* __restrict__ h_states, const float* __restrict__ W_sp,
    const float* __restrict__ b_sp, __bf16* __restrict__ X) {
  __shared__ float Wsp_s[16 * 512];
  __shared__ float relS[64 * 16];
  const int s = blockIdx.x >> 6;
  const int i = blockIdx.x & 63;
  const int tid = threadIdx.x;

  // stage W_sp (32 KB) as float4
  const float4* w4  = (const float4*)W_sp;
  float4*       ws4 = (float4*)Wsp_s;
  for (int idx = tid; idx < 2048; idx += 256) ws4[idx] = w4[idx];

  // rel[j][t*2+k] = obs[s,j,t,k] - obs[s,i,t,k]; obs[s,p,t,k] = traj[t,s*64+p,k]
  for (int idx = tid; idx < 64 * 16; idx += 256) {
    int j = idx >> 4, q = idx & 15, t = q >> 1, k = q & 1;
    float oj = traj[(t * 2048 + s * 64 + j) * 2 + k];
    float oi = traj[(t * 2048 + s * 64 + i) * 2 + k];
    relS[idx] = oj - oi;
  }
  __syncthreads();

  // 4 threads per j, 128 channels each; emit packed b128 stores (8 bf16)
  const int j  = tid >> 2;
  const int c0 = (tid & 3) * 128;
  float rl[16];
#pragma unroll
  for (int k = 0; k < 16; ++k) rl[k] = relS[j * 16 + k];
  const long row    = (long)(s * PP + i * 64 + j);
  const long twbase = row * 16;  // traj_weight[s, p, :, :] = 2*8 floats
  for (int cc = 0; cc < 128; cc += 8) {
    Pack8 pk;
#pragma unroll
    for (int u = 0; u < 8; ++u) {
      int c = c0 + cc + u;
      float e = b_sp[c];
#pragma unroll
      for (int k = 0; k < 16; ++k) e += rl[k] * Wsp_s[k * 512 + c];
      int t = c >> 6, ee = c & 63;
      float w = traj_weight[twbase + (ee & 1) * 8 + t];
      pk.h[u] = (__bf16)(e * w);
    }
    *(uint4*)(X + row * PRE + c0 + cc) = pk.u4;
  }

  // hidden of ped j: 64 rows x 64 cols, 8 bf16 per store
  for (int idx = tid; idx < 64 * 8; idx += 256) {
    int jj = idx >> 3, g = (idx & 7) * 8;
    const float4 a = *(const float4*)(h_states + (s * 64 + jj) * 64 + g);
    const float4 b = *(const float4*)(h_states + (s * 64 + jj) * 64 + g + 4);
    Pack8 pk;
    pk.h[0] = (__bf16)a.x; pk.h[1] = (__bf16)a.y;
    pk.h[2] = (__bf16)a.z; pk.h[3] = (__bf16)a.w;
    pk.h[4] = (__bf16)b.x; pk.h[5] = (__bf16)b.y;
    pk.h[6] = (__bf16)b.z; pk.h[7] = (__bf16)b.w;
    *(uint4*)(X + (long)(s * PP + i * 64 + jj) * PRE + 512 + g) = pk.u4;
  }
}

// ---------------------------------------------------------------------------
// GEMM1: y1 = X[131072,576] @ W1 + b1 (bf16 out, f32 accum), stats1 epilogue.
// Tile 128x128, BLOCK_K=32, double-buffered (compile-time ping-pong via 2x
// manual unroll so all DS offsets are immediates), LDS-bounced C store.
// ---------------------------------------------------------------------------
__global__ __launch_bounds__(256) void gemm1_kernel(
    const __bf16* __restrict__ X, const __bf16* __restrict__ W1t,
    const float* __restrict__ b1, __bf16* __restrict__ y1,
    float* __restrict__ stats1) {
  __shared__ __align__(16) char smem[32768];  // A0|B0|A1|B1 ; C aliases all
  __shared__ float csum[128], csq[128];

  const int tid = threadIdx.x, lane = tid & 31, wave = tid >> 5;
  const int wm = wave & 3, wn = wave >> 2;
  const int tm = blockIdx.y, tn = blockIdx.x;
  const int seq = tm >> 5;           // 32 row-tiles per sequence
  const long rowg0 = (long)tm * 128;
  const int  colg0 = tn * 128;

  v8f acc[2][4];
#pragma unroll
  for (int a = 0; a < 2; ++a)
#pragma unroll
    for (int b = 0; b < 4; ++b)
#pragma unroll
      for (int v = 0; v < 8; ++v) acc[a][b][v] = 0.f;

  for (int c = tid; c < 128; c += 256) { csum[c] = 0.f; csq[c] = 0.f; }

  auto stage = [&](int kt2, char* half) {
    __bf16* sA = (__bf16*)half;
    __bf16* sB = (__bf16*)(half + 8192);
#pragma unroll
    for (int it = 0; it < 2; ++it) {      // A: 128 rows x 32 k
      int idx = tid + it * 256;
      int r = idx >> 2, q = idx & 3;
      uint4 d = *(const uint4*)(X + (rowg0 + r) * PRE + kt2 * 32 + q * 8);
      *((uint4*)(sA + r * 32 + q * 8)) = d;
    }
#pragma unroll
    for (int it = 0; it < 2; ++it) {      // B: 128 n x 32 k ([N][K] src)
      int idx = tid + it * 256;
      int n = idx >> 2, q = idx & 3;
      uint4 d = *(const uint4*)(W1t + (long)(colg0 + n) * PRE + kt2 * 32 + q * 8);
      *((uint4*)(sB + n * 32 + q * 8)) = d;
    }
  };

  auto compute = [&](char* cur) {
    unsigned* Au = (unsigned*)cur;
    unsigned* Bu = (unsigned*)(cur + 8192);
    FragU af[2], bf4[4];
    const int hs = (lane >> 4) * 4;  // A: lanes>=16 take K=8..15 / 24..31
#pragma unroll
    for (int mt = 0; mt < 2; ++mt) {
      int r = wm * 32 + mt * 16 + (lane & 15);
#pragma unroll
      for (int v = 0; v < 4; ++v) {
        af[mt].u[v]     = Au[r * 16 + hs + v];
        af[mt].u[v + 4] = Au[r * 16 + 8 + hs + v];
      }
    }
    const int ks = (lane >> 4) * 4;  // B: lanes>=16 take K=8..15 / 24..31
#pragma unroll
    for (int nt = 0; nt < 4; ++nt) {
      int n = wn * 64 + nt * 16 + (lane & 15);
#pragma unroll
      for (int v = 0; v < 4; ++v) {
        bf4[nt].u[v]     = Bu[n * 16 + ks + v];
        bf4[nt].u[v + 4] = Bu[n * 16 + 8 + ks + v];
      }
    }
#pragma unroll
    for (int mt = 0; mt < 2; ++mt)
#pragma unroll
      for (int nt = 0; nt < 4; ++nt)
        acc[mt][nt] = __builtin_amdgcn_wmma_f32_16x16x32_bf16(
            false, af[mt].v, false, bf4[nt].v, (short)0, acc[mt][nt], false, false);
  };

  const int KT = PRE / 32;  // 18 (even)
  stage(0, smem);
  for (int kt = 0; kt < KT; kt += 2) {
    __syncthreads();
    stage(kt + 1, smem + 16384);     // kt+1 <= KT-1 always (KT even)
    compute(smem);
    __syncthreads();
    if (kt + 2 < KT) stage(kt + 2, smem);
    compute(smem + 16384);
  }
  __syncthreads();   // staging dead; smem becomes C tile

  // epilogue: bias, stats into LDS, C tile into LDS (bf16)
  __bf16* Cs = (__bf16*)smem;  // [128][128]
  const int rhalf = (lane >> 4) * 8;
#pragma unroll
  for (int mt = 0; mt < 2; ++mt) {
#pragma unroll
    for (int nt = 0; nt < 4; ++nt) {
      int col_l = wn * 64 + nt * 16 + (lane & 15);
      float bias = b1[colg0 + col_l];
#pragma unroll
      for (int v = 0; v < 8; ++v) {
        int row_l = wm * 32 + mt * 16 + v + rhalf;
        float y = acc[mt][nt][v] + bias;
        Cs[row_l * 128 + col_l] = (__bf16)y;
        atomicAdd(&csum[col_l], y);
        atomicAdd(&csq[col_l],  y * y);
      }
    }
  }
  __syncthreads();
  // coalesced b128 store of C tile
#pragma unroll
  for (int it = 0; it < 8; ++it) {
    int idx = tid + it * 256;
    int r = idx >> 4, g = (idx & 15) * 8;
    *(uint4*)(y1 + (rowg0 + r) * D1 + colg0 + g) = *(uint4*)(Cs + r * 128 + g);
  }
  for (int c = tid; c < 128; c += 256) {
    atomicAdd(&stats1[seq * D1 + colg0 + c], csum[c]);
    atomicAdd(&stats1[S_ * D1 + seq * D1 + colg0 + c], csq[c]);
  }
}

// ---------------------------------------------------------------------------
// GEMM2: y2 = relu(bn1(y1)) @ W2 + b2, stats2 epilogue.
// BN1+ReLU fused into A staging using per-sequence coefficients.
// ---------------------------------------------------------------------------
__global__ __launch_bounds__(256) void gemm2_kernel(
    const __bf16* __restrict__ y1, const __bf16* __restrict__ W2t,
    const float* __restrict__ b2, const float* __restrict__ g1,
    const float* __restrict__ be1, const float* __restrict__ stats1,
    __bf16* __restrict__ y2, float* __restrict__ stats2) {
  __shared__ __align__(16) char smem[32768];
  __shared__ float csum[128], csq[128];
  __shared__ float sc1[512], sh1[512];

  const int tid = threadIdx.x, lane = tid & 31, wave = tid >> 5;
  const int wm = wave & 3, wn = wave >> 2;
  const int tm = blockIdx.y, tn = blockIdx.x;
  const int seq = tm >> 5;
  const long rowg0 = (long)tm * 128;
  const int  colg0 = tn * 128;

  // BN1 coefficients for this sequence
  const float inv = 1.0f / 4096.0f;
  for (int c = tid; c < 512; c += 256) {
    float sum  = stats1[seq * D1 + c];
    float sq   = stats1[S_ * D1 + seq * D1 + c];
    float mean = sum * inv;
    float var  = sq * inv - mean * mean;
    float rs   = rsqrtf(var + EPSBN);
    float sc   = g1[c] * rs;
    sc1[c] = sc;
    sh1[c] = be1[c] - mean * sc;
  }
  for (int c = tid; c < 128; c += 256) { csum[c] = 0.f; csq[c] = 0.f; }
  __syncthreads();  // sc1/sh1 ready before first A staging

  v8f acc[2][4];
#pragma unroll
  for (int a = 0; a < 2; ++a)
#pragma unroll
    for (int b = 0; b < 4; ++b)
#pragma unroll
      for (int v = 0; v < 8; ++v) acc[a][b][v] = 0.f;

  auto stage = [&](int kt2, char* half) {
    __bf16* sA = (__bf16*)half;
    __bf16* sB = (__bf16*)(half + 8192);
#pragma unroll
    for (int it = 0; it < 2; ++it) {      // A + fused BN1/ReLU
      int idx = tid + it * 256;
      int r = idx >> 2, q = idx & 3;
      int gk = kt2 * 32 + q * 8;
      Pack8 p; p.u4 = *(const uint4*)(y1 + (rowg0 + r) * D1 + gk);
#pragma unroll
      for (int u = 0; u < 8; ++u) {
        float v = (float)p.h[u];
        v = fmaxf(v * sc1[gk + u] + sh1[gk + u], 0.f);
        p.h[u] = (__bf16)v;
      }
      *((uint4*)(sA + r * 32 + q * 8)) = p.u4;
    }
#pragma unroll
    for (int it = 0; it < 2; ++it) {      // B ([N][K] src)
      int idx = tid + it * 256;
      int n = idx >> 2, q = idx & 3;
      uint4 d = *(const uint4*)(W2t + (long)(colg0 + n) * D1 + kt2 * 32 + q * 8);
      *((uint4*)(sB + n * 32 + q * 8)) = d;
    }
  };

  auto compute = [&](char* cur) {
    unsigned* Au = (unsigned*)cur;
    unsigned* Bu = (unsigned*)(cur + 8192);
    FragU af[2], bf4[4];
    const int hs = (lane >> 4) * 4;
#pragma unroll
    for (int mt = 0; mt < 2; ++mt) {
      int r = wm * 32 + mt * 16 + (lane & 15);
#pragma unroll
      for (int v = 0; v < 4; ++v) {
        af[mt].u[v]     = Au[r * 16 + hs + v];
        af[mt].u[v + 4] = Au[r * 16 + 8 + hs + v];
      }
    }
    const int ks = (lane >> 4) * 4;
#pragma unroll
    for (int nt = 0; nt < 4; ++nt) {
      int n = wn * 64 + nt * 16 + (lane & 15);
#pragma unroll
      for (int v = 0; v < 4; ++v) {
        bf4[nt].u[v]     = Bu[n * 16 + ks + v];
        bf4[nt].u[v + 4] = Bu[n * 16 + 8 + ks + v];
      }
    }
#pragma unroll
    for (int mt = 0; mt < 2; ++mt)
#pragma unroll
      for (int nt = 0; nt < 4; ++nt)
        acc[mt][nt] = __builtin_amdgcn_wmma_f32_16x16x32_bf16(
            false, af[mt].v, false, bf4[nt].v, (short)0, acc[mt][nt], false, false);
  };

  const int KT = D1 / 32;  // 16 (even)
  stage(0, smem);
  for (int kt = 0; kt < KT; kt += 2) {
    __syncthreads();
    stage(kt + 1, smem + 16384);
    compute(smem);
    __syncthreads();
    if (kt + 2 < KT) stage(kt + 2, smem);
    compute(smem + 16384);
  }
  __syncthreads();

  __bf16* Cs = (__bf16*)smem;
  const int rhalf = (lane >> 4) * 8;
#pragma unroll
  for (int mt = 0; mt < 2; ++mt) {
#pragma unroll
    for (int nt = 0; nt < 4; ++nt) {
      int col_l = wn * 64 + nt * 16 + (lane & 15);
      float bias = b2[colg0 + col_l];
#pragma unroll
      for (int v = 0; v < 8; ++v) {
        int row_l = wm * 32 + mt * 16 + v + rhalf;
        float y = acc[mt][nt][v] + bias;
        Cs[row_l * 128 + col_l] = (__bf16)y;
        atomicAdd(&csum[col_l], y);
        atomicAdd(&csq[col_l],  y * y);
      }
    }
  }
  __syncthreads();
#pragma unroll
  for (int it = 0; it < 8; ++it) {
    int idx = tid + it * 256;
    int r = idx >> 4, g = (idx & 15) * 8;
    *(uint4*)(y2 + (rowg0 + r) * D2 + colg0 + g) = *(uint4*)(Cs + r * 128 + g);
  }
  for (int c = tid; c < 128; c += 256) {
    atomicAdd(&stats2[seq * D2 + colg0 + c], csum[c]);
    atomicAdd(&stats2[S_ * D2 + seq * D2 + colg0 + c], csq[c]);
  }
}

// ---------------------------------------------------------------------------
// BN2 + ReLU + max-pool over j.  One workgroup per (s, i); thread owns 4 ch.
// ---------------------------------------------------------------------------
__global__ __launch_bounds__(256) void pool_kernel(
    const __bf16* __restrict__ y2, const float* __restrict__ stats2,
    const float* __restrict__ g2, const float* __restrict__ be2,
    float* __restrict__ out) {
  const int s = blockIdx.x >> 6, i = blockIdx.x & 63;
  const int c0 = threadIdx.x * 4;
  const float inv = 1.0f / 4096.0f;
  float sc[4], sh[4], mx[4];
#pragma unroll
  for (int p = 0; p < 4; ++p) {
    int c = c0 + p;
    float sum  = stats2[s * D2 + c];
    float sq   = stats2[S_ * D2 + s * D2 + c];
    float mean = sum * inv;
    float var  = sq * inv - mean * mean;
    float rs   = rsqrtf(var + EPSBN);
    sc[p] = g2[c] * rs;
    sh[p] = be2[c] - mean * sc[p];
    mx[p] = 0.f;  // relu output >= 0, so 0 is a valid identity for max
  }
  const long base = (long)(s * PP + i * 64) * D2 + c0;
  for (int j = 0; j < 64; ++j) {
    Pack4 p4; p4.u2 = *(const uint2*)(y2 + base + (long)j * D2);
#pragma unroll
    for (int p = 0; p < 4; ++p) {
      float v = fmaxf((float)p4.h[p] * sc[p] + sh[p], 0.f);
      mx[p] = fmaxf(mx[p], v);
    }
  }
  float4 o; o.x = mx[0]; o.y = mx[1]; o.z = mx[2]; o.w = mx[3];
  *(float4*)(out + (long)(s * 64 + i) * D2 + c0) = o;
}

// ---------------------------------------------------------------------------
extern "C" void kernel_launch(void* const* d_in, const int* in_sizes, int n_in,
                              void* d_out, int out_size, void* d_ws, size_t ws_size,
                              hipStream_t stream) {
  const float* h_states    = (const float*)d_in[0];
  const float* traj        = (const float*)d_in[3];
  const float* traj_weight = (const float*)d_in[4];
  const float* W_sp        = (const float*)d_in[6];
  const float* b_sp        = (const float*)d_in[7];
  const float* W1          = (const float*)d_in[8];
  const float* b1          = (const float*)d_in[9];
  const float* g1          = (const float*)d_in[10];
  const float* be1         = (const float*)d_in[11];
  const float* W2          = (const float*)d_in[12];
  const float* b2          = (const float*)d_in[13];
  const float* g2          = (const float*)d_in[14];
  const float* be2         = (const float*)d_in[15];
  float* out = (float*)d_out;

  char* ws = (char*)d_ws;
  size_t off = 0;
  auto alloc = [&](size_t bytes) -> void* {
    void* p = ws + off;
    off += (bytes + 255) & ~(size_t)255;
    return p;
  };
  __bf16* X      = (__bf16*)alloc((size_t)ROWS * PRE * 2);
  __bf16* y1     = (__bf16*)alloc((size_t)ROWS * D1 * 2);
  __bf16* y2     = (__bf16*)alloc((size_t)ROWS * D2 * 2);
  __bf16* W1t    = (__bf16*)alloc((size_t)PRE * D1 * 2);   // [N=512][K=576]
  __bf16* W2t    = (__bf16*)alloc((size_t)D1 * D2 * 2);    // [N=1024][K=512]
  float*  stats1 = (float*)alloc((size_t)2 * S_ * D1 * 4);
  float*  stats2 = (float*)alloc((size_t)2 * S_ * D2 * 4);

  hipMemsetAsync(stats1, 0, (size_t)2 * S_ * D1 * 4, stream);
  hipMemsetAsync(stats2, 0, (size_t)2 * S_ * D2 * 4, stream);

  cvt_wt_kernel<<<(PRE * D1 + 255) / 256, 256, 0, stream>>>(W1, W1t, PRE, D1);
  cvt_wt_kernel<<<(D1 * D2 + 255) / 256, 256, 0, stream>>>(W2, W2t, D1, D2);
  build_x_kernel<<<S_ * P_, 256, 0, stream>>>(traj, traj_weight, h_states, W_sp, b_sp, X);
  gemm1_kernel<<<dim3(D1 / 128, ROWS / 128), 256, 0, stream>>>(X, W1t, b1, y1, stats1);
  gemm2_kernel<<<dim3(D2 / 128, ROWS / 128), 256, 0, stream>>>(y1, W2t, b2, g1, be1,
                                                               stats1, y2, stats2);
  pool_kernel<<<S_ * P_, 256, 0, stream>>>(y2, stats2, g2, be2, out);
}
